// GCNLayer_48473000903495
// MI455X (gfx1250) — matrix operand order repaired and checked by reading, not compiled, not run on hardware
//
#include <hip/hip_runtime.h>
#include <hip/hip_bf16.h>

#define NROWS 8192
#define DDIM  256
#define LDA   40
#define LDB   40

typedef __attribute__((ext_vector_type(16))) __bf16         bf16x16;
typedef __attribute__((ext_vector_type(8)))  float          f32x8;
typedef __attribute__((ext_vector_type(4)))  unsigned int   uint4v;
typedef __attribute__((ext_vector_type(4)))  int            int4v;
typedef __attribute__((ext_vector_type(4)))  float          float4v;
typedef __attribute__((ext_vector_type(4)))  unsigned short ushort4v;

struct Frag32B { uint4v lo, hi; };

#if __has_builtin(__builtin_amdgcn_global_load_async_to_lds_b128) && \
    __has_builtin(__builtin_amdgcn_s_wait_asynccnt)
#define HAVE_ASYNC_LDS 1
#else
#define HAVE_ASYNC_LDS 0
#endif

typedef __attribute__((address_space(1))) int4v gbl_int4v;
typedef __attribute__((address_space(3))) int4v lds_int4v;

static __device__ __forceinline__ gbl_int4v* gbl_cast(const void* p) {
  // flat global address == AS1 address
  return (gbl_int4v*)(unsigned long long)p;
}
static __device__ __forceinline__ lds_int4v* lds_cast(void* p) {
  // low 32 bits of a flat LDS address are the LDS byte offset
  return (lds_int4v*)(unsigned int)(unsigned long long)p;
}

static __device__ __forceinline__ unsigned short f2bf(float f) {
  __bf16 h = (__bf16)f;
  return __builtin_bit_cast(unsigned short, h);
}

static __device__ __forceinline__ void pipeline_sync() {
#if HAVE_ASYNC_LDS
  __builtin_amdgcn_s_wait_asynccnt(0);
#endif
  __syncthreads();
}

// copy 16B per lane, global -> LDS (async if available)
static __device__ __forceinline__ void copy16_g2l(unsigned short* dst,
                                                  const unsigned short* src) {
#if HAVE_ASYNC_LDS
  __builtin_amdgcn_global_load_async_to_lds_b128(gbl_cast(src), lds_cast(dst), 0, 0);
#else
  *(uint4v*)dst = *(const uint4v*)src;
#endif
}

// ---------------- Kernel 1: inv_deg[i] = 1 / rowsum(A[i,:]) ----------------
__global__ void degree_kernel(const float* __restrict__ adj,
                              float* __restrict__ invdeg) {
  __shared__ float red[256];
  const int row = blockIdx.x;
  const int t = threadIdx.x;
  const float4v* rp = (const float4v*)(adj + (size_t)row * NROWS);
  float s = 0.f;
#pragma unroll
  for (int i = 0; i < 8; ++i) {
    float4v v = rp[t + i * 256];
    s += v.x + v.y + v.z + v.w;
  }
  red[t] = s;
  __syncthreads();
  for (int off = 128; off > 0; off >>= 1) {
    if (t < off) red[t] += red[t + off];
    __syncthreads();
  }
  if (t == 0) invdeg[row] = 1.0f / red[0];
}

// ------- Kernel 2: xsT[n][k] = bf16( x[k][n] * inv_deg[k] )  (64x64 tiles) -------
__global__ void xst_kernel(const float* __restrict__ x,
                           const float* __restrict__ invdeg,
                           unsigned short* __restrict__ xsT) {
  __shared__ unsigned short tile[64][66];
  const int k0 = blockIdx.x * 64;
  const int n0 = blockIdx.y * 64;
  const int t = threadIdx.x;
  const int c4 = t & 15;
  const int r  = t >> 4;
#pragma unroll
  for (int p = 0; p < 4; ++p) {
    int k = r + p * 16;
    float sc = invdeg[k0 + k];
    float4v v = *(const float4v*)(x + (size_t)(k0 + k) * DDIM + n0 + c4 * 4);
    tile[k][c4 * 4 + 0] = f2bf(v.x * sc);
    tile[k][c4 * 4 + 1] = f2bf(v.y * sc);
    tile[k][c4 * 4 + 2] = f2bf(v.z * sc);
    tile[k][c4 * 4 + 3] = f2bf(v.w * sc);
  }
  __syncthreads();
  const int n  = t >> 2;
  const int kq = t & 3;
  unsigned int* out = (unsigned int*)xsT;
#pragma unroll
  for (int j = 0; j < 8; ++j) {
    int k = kq * 16 + j * 2;
    unsigned int u = (unsigned int)tile[k][n] | ((unsigned int)tile[k + 1][n] << 16);
    out[(size_t)(n0 + n) * (NROWS / 2) + ((k0 + k) >> 1)] = u;
  }
}

// ---------------- Kernel 2b: WT[n][k] = bf16(W[k][n]) ----------------
__global__ void wt_kernel(const float* __restrict__ w,
                          unsigned short* __restrict__ wt) {
  const int k = blockIdx.x;
  const int n = threadIdx.x;
  wt[n * DDIM + k] = f2bf(w[k * DDIM + n]);
}

// ---- staging helpers ----
static __device__ __forceinline__ void stage_agg_A(unsigned short* ldsAbuf,
                                                   const float* adj,
                                                   int rowBase, int kc, int t) {
#pragma unroll
  for (int i = 0; i < 2; ++i) {
    int f = t + i * 256;
    int r  = f >> 3;
    int c4 = f & 7;
    float4v v = *(const float4v*)(adj + (size_t)(rowBase + r) * NROWS + kc + c4 * 4);
    ushort4v p;
    p.x = f2bf(v.x); p.y = f2bf(v.y); p.z = f2bf(v.z); p.w = f2bf(v.w);
    *(ushort4v*)&ldsAbuf[r * LDA + c4 * 4] = p;
  }
}

static __device__ __forceinline__ void stage_B(unsigned short* ldsBbuf,
                                               const unsigned short* src,
                                               int srcStride, int kc, int t) {
  const unsigned short* s = src + (size_t)t * srcStride + kc;
  unsigned short* d = &ldsBbuf[t * LDB];
#pragma unroll
  for (int j = 0; j < 4; ++j) copy16_g2l(d + j * 8, s + j * 8);
}

static __device__ __forceinline__ void stage_copy_A(unsigned short* ldsAbuf,
                                                    const unsigned short* src,
                                                    int srcStride, int rowBase,
                                                    int kc, int t) {
  int r   = t >> 2;
  int c16 = t & 3;
  copy16_g2l(&ldsAbuf[r * LDA + c16 * 8],
             src + (size_t)(rowBase + r) * srcStride + kc + c16 * 8);
}

// -------- Kernel 3: h = diag(invdeg) * (A @ xs), output bf16 [N][D] --------
__global__ void __launch_bounds__(256, 1)
agg_kernel(const float* __restrict__ adj,
           const unsigned short* __restrict__ xsT,
           const float* __restrict__ invdeg,
           unsigned short* __restrict__ hbf) {
  __shared__ unsigned short ldsA[2][64 * LDA];
  __shared__ unsigned short ldsB[2][256 * LDB];
  const int t = threadIdx.x;
  const int lane = t & 31;
  const int wave = t >> 5;
  const int rtile = wave >> 1;
  const int chalf = wave & 1;
  const int rowBase = blockIdx.x * 64;

  f32x8 acc[8];
#pragma unroll
  for (int i = 0; i < 8; ++i)
#pragma unroll
    for (int v = 0; v < 8; ++v) acc[i][v] = 0.0f;

  const int aRow = rtile * 16 + (lane & 15);
  const int aK0  = (lane >> 4) * 8;
  const int bN   = lane & 15;
  const int bK0  = (lane >> 4) * 8;

  // prologue: stage tile 0 into buffer 0
  stage_agg_A(ldsA[0], adj, rowBase, 0, t);
  stage_B(ldsB[0], xsT, NROWS, 0, t);
  pipeline_sync();

  for (int kc = 0; kc < NROWS; kc += 32) {
    const int cur = (kc >> 5) & 1;
    if (kc + 32 < NROWS) {
      stage_agg_A(ldsA[1 - cur], adj, rowBase, kc + 32, t);
      stage_B(ldsB[1 - cur], xsT, NROWS, kc + 32, t);
    }

    // load A fragment + all 8 B fragments, then issue WMMAs back-to-back
    Frag32B fa;
    const unsigned short* pa = &ldsA[cur][aRow * LDA + aK0];
    fa.lo = *(const uint4v*)pa;
    fa.hi = *(const uint4v*)(pa + 16);
    bf16x16 a = __builtin_bit_cast(bf16x16, fa);

    bf16x16 bfrag[8];
#pragma unroll
    for (int ct = 0; ct < 8; ++ct) {
      Frag32B fb;
      const unsigned short* pb =
          &ldsB[cur][(chalf * 128 + ct * 16 + bN) * LDB + bK0];
      fb.lo = *(const uint4v*)pb;
      fb.hi = *(const uint4v*)(pb + 16);
      bfrag[ct] = __builtin_bit_cast(bf16x16, fb);
    }
#pragma unroll
    for (int ct = 0; ct < 8; ++ct)
      acc[ct] = __builtin_amdgcn_wmma_f32_16x16x32_bf16(
          false, a, false, bfrag[ct], (short)0, acc[ct], false, false);

    pipeline_sync();
  }

  // epilogue: row-scale by inv_deg, store bf16
  const int rBase = rowBase + rtile * 16 + (lane >> 4) * 8;
  float invd[8];
#pragma unroll
  for (int v = 0; v < 8; ++v) invd[v] = invdeg[rBase + v];
#pragma unroll
  for (int ct = 0; ct < 8; ++ct) {
    int col = chalf * 128 + ct * 16 + (lane & 15);
#pragma unroll
    for (int v = 0; v < 8; ++v)
      hbf[(size_t)(rBase + v) * DDIM + col] = f2bf(acc[ct][v] * invd[v]);
  }
}

// -------- Kernel 4: out = relu(h @ W + bias), f32 output --------
__global__ void __launch_bounds__(256, 1)
out_kernel(const unsigned short* __restrict__ hbf,
           const unsigned short* __restrict__ wt,
           const float* __restrict__ bias,
           float* __restrict__ out) {
  __shared__ unsigned short ldsA[2][64 * LDA];
  __shared__ unsigned short ldsB[2][256 * LDB];
  const int t = threadIdx.x;
  const int lane = t & 31;
  const int wave = t >> 5;
  const int rtile = wave >> 1;
  const int chalf = wave & 1;
  const int rowBase = blockIdx.x * 64;

  f32x8 acc[8];
#pragma unroll
  for (int i = 0; i < 8; ++i)
#pragma unroll
    for (int v = 0; v < 8; ++v) acc[i][v] = 0.0f;

  const int aRow = rtile * 16 + (lane & 15);
  const int aK0  = (lane >> 4) * 8;
  const int bN   = lane & 15;
  const int bK0  = (lane >> 4) * 8;

  stage_copy_A(ldsA[0], hbf, DDIM, rowBase, 0, t);
  stage_B(ldsB[0], wt, DDIM, 0, t);
  pipeline_sync();

  for (int kc = 0; kc < DDIM; kc += 32) {
    const int cur = (kc >> 5) & 1;
    if (kc + 32 < DDIM) {
      stage_copy_A(ldsA[1 - cur], hbf, DDIM, rowBase, kc + 32, t);
      stage_B(ldsB[1 - cur], wt, DDIM, kc + 32, t);
    }

    Frag32B fa;
    const unsigned short* pa = &ldsA[cur][aRow * LDA + aK0];
    fa.lo = *(const uint4v*)pa;
    fa.hi = *(const uint4v*)(pa + 16);
    bf16x16 a = __builtin_bit_cast(bf16x16, fa);

    bf16x16 bfrag[8];
#pragma unroll
    for (int ct = 0; ct < 8; ++ct) {
      Frag32B fb;
      const unsigned short* pb =
          &ldsB[cur][(chalf * 128 + ct * 16 + bN) * LDB + bK0];
      fb.lo = *(const uint4v*)pb;
      fb.hi = *(const uint4v*)(pb + 16);
      bfrag[ct] = __builtin_bit_cast(bf16x16, fb);
    }
#pragma unroll
    for (int ct = 0; ct < 8; ++ct)
      acc[ct] = __builtin_amdgcn_wmma_f32_16x16x32_bf16(
          false, a, false, bfrag[ct], (short)0, acc[ct], false, false);

    pipeline_sync();
  }

  const int rBase = rowBase + rtile * 16 + (lane >> 4) * 8;
#pragma unroll
  for (int ct = 0; ct < 8; ++ct) {
    int col = chalf * 128 + ct * 16 + (lane & 15);
    float b = bias[col];
#pragma unroll
    for (int v = 0; v < 8; ++v) {
      float val = acc[ct][v] + b;
      out[(size_t)(rBase + v) * DDIM + col] = val > 0.0f ? val : 0.0f;
    }
  }
}

extern "C" void kernel_launch(void* const* d_in, const int* in_sizes, int n_in,
                              void* d_out, int out_size, void* d_ws, size_t ws_size,
                              hipStream_t stream) {
  (void)in_sizes; (void)n_in; (void)out_size; (void)ws_size;
  const float* x    = (const float*)d_in[0];
  const float* adj  = (const float*)d_in[1];
  const float* w    = (const float*)d_in[2];
  const float* bias = (const float*)d_in[3];
  float* out = (float*)d_out;

  char* ws = (char*)d_ws;
  float*          invdeg = (float*)ws;                                       // 32 KB
  unsigned short* xsT    = (unsigned short*)(ws + 32768);                    // 4 MB
  unsigned short* wt     = (unsigned short*)(ws + 32768 + 4194304);          // 128 KB
  unsigned short* hbf    = (unsigned short*)(ws + 32768 + 4194304 + 131072); // 4 MB

  degree_kernel<<<dim3(NROWS), 256, 0, stream>>>(adj, invdeg);
  xst_kernel<<<dim3(NROWS / 64, DDIM / 64), 256, 0, stream>>>(x, invdeg, xsT);
  wt_kernel<<<dim3(DDIM), 256, 0, stream>>>(w, wt);
  agg_kernel<<<dim3(NROWS / 64), 256, 0, stream>>>(adj, xsT, invdeg, hbf);
  out_kernel<<<dim3(NROWS / 64), 256, 0, stream>>>(hbf, wt, bias, out);
}
